// LinearSelfAttention_57346403336825
// MI455X (gfx1250) — compile-verified, matrix operand order checked by reading, not run on hardware
//
#include <hip/hip_runtime.h>
#include <hip/hip_bf16.h>

typedef __bf16 bf16_t;
typedef __attribute__((ext_vector_type(16))) __bf16 v16bf;
typedef __attribute__((ext_vector_type(8)))  __bf16 v8bf;
typedef __attribute__((ext_vector_type(8)))  float  v8f;

#define E_DIM 256
#define N_TOK 1024
#define BP    128            // B*P = 32*4
#define TOKENS (BP * N_TOK)  // 131072

// ---------------------------------------------------------------------------
// k0: pack w_q column contiguously; convert W_v / W_o to bf16 in the exact
// per-lane WMMA B-fragment layout:
//   block (kt,nt): 32 lanes x 16 bf16.  lane = 16*khalf + col_in_tile.
//   elem i<8  -> K = kt*32 + base0 + i        (base0 = 8*khalf)
//   elem i>=8 -> K = kt*32 + base0 + 16 + i-8
// ---------------------------------------------------------------------------
__global__ __launch_bounds__(256) void k0_prep(
    const float* __restrict__ W_qkv, const float* __restrict__ W_o,
    float* __restrict__ wq_ws, unsigned short* __restrict__ wv_pre,
    unsigned short* __restrict__ wo_pre)
{
  int idx = blockIdx.x * 256 + threadIdx.x;
  if (idx < 256) wq_ws[idx] = W_qkv[idx * 513];   // q column (stride 513)

  int e    = idx & 65535;
  int i    = e & 15;
  int lane = (e >> 4) & 31;
  int blk  = e >> 9;
  int nt   = blk & 15;
  int kt   = blk >> 4;
  int col  = nt * 16 + (lane & 15);
  int b0   = (lane < 16) ? 0 : 8;
  int kloc = (i < 8) ? (b0 + i) : (b0 + 16 + (i - 8));
  int K    = kt * 32 + kloc;

  if (idx < 65536) {
    ((bf16_t*)wv_pre)[e] = (bf16_t)W_qkv[K * 513 + 257 + col];   // W_v slice
  } else if (idx < 131072) {
    ((bf16_t*)wo_pre)[e] = (bf16_t)W_o[K * 256 + col];
  }
}

// ---------------------------------------------------------------------------
// k1: q_n = x_n . w_q + b_q   (one wave per token, 8 tokens per block)
// ---------------------------------------------------------------------------
__global__ __launch_bounds__(256) void k1_q(
    const float* __restrict__ x, const float* __restrict__ wq_ws,
    const float* __restrict__ b_qkv, float* __restrict__ q_ws)
{
  __shared__ float wqs[256];
  int tid = threadIdx.x;
  wqs[tid] = wq_ws[tid];
  __syncthreads();

  int wave = tid >> 5, lane = tid & 31;
  long token = (long)blockIdx.x * 8 + wave;
  const float* xp = x + token * E_DIM + lane * 8;
  const float* wp = wqs + lane * 8;
  float s = 0.0f;
#pragma unroll
  for (int j = 0; j < 8; ++j) s += xp[j] * wp[j];
#pragma unroll
  for (int off = 16; off; off >>= 1) s += __shfl_xor(s, off, 32);
  if (lane == 0) q_ws[token] = s + b_qkv[0];
}

// ---------------------------------------------------------------------------
// k2: per (b,p): softmax over N of q, xbar = sum_n s_n * x_n,
//     cv = xbar @ W_k + b_k   (since sum s_n = 1, bias folds exactly)
// ---------------------------------------------------------------------------
__global__ __launch_bounds__(256) void k2_cv(
    const float* __restrict__ x, const float* __restrict__ W_qkv,
    const float* __restrict__ b_qkv, const float* __restrict__ q_ws,
    float* __restrict__ cv_ws)
{
  __shared__ float sprob[N_TOK];
  __shared__ float red[8];
  __shared__ float xbar[E_DIM];

  int bp = blockIdx.x, tid = threadIdx.x;
  int lane = tid & 31, wid = tid >> 5;

  const float* qp = q_ws + (long)bp * N_TOK;
  float qv[4];
#pragma unroll
  for (int j = 0; j < 4; ++j) qv[j] = qp[tid + j * 256];

  float m = fmaxf(fmaxf(qv[0], qv[1]), fmaxf(qv[2], qv[3]));
#pragma unroll
  for (int off = 16; off; off >>= 1) m = fmaxf(m, __shfl_xor(m, off, 32));
  if (lane == 0) red[wid] = m;
  __syncthreads();
  float gm = red[0];
#pragma unroll
  for (int w = 1; w < 8; ++w) gm = fmaxf(gm, red[w]);
  __syncthreads();

  float e0 = expf(qv[0] - gm), e1 = expf(qv[1] - gm);
  float e2 = expf(qv[2] - gm), e3 = expf(qv[3] - gm);
  float ls = e0 + e1 + e2 + e3;
#pragma unroll
  for (int off = 16; off; off >>= 1) ls += __shfl_xor(ls, off, 32);
  if (lane == 0) red[wid] = ls;
  __syncthreads();
  float denom = 0.0f;
#pragma unroll
  for (int w = 0; w < 8; ++w) denom += red[w];
  float inv = 1.0f / denom;
  sprob[tid]       = e0 * inv;
  sprob[tid + 256] = e1 * inv;
  sprob[tid + 512] = e2 * inv;
  sprob[tid + 768] = e3 * inv;
  __syncthreads();

  // xbar[d] = sum_n s_n * x[bp,n,d]   (coalesced: thread = channel d)
  const float* xp = x + (long)bp * N_TOK * E_DIM + tid;
  float acc = 0.0f;
  for (int n = 0; n < N_TOK; ++n) acc += sprob[n] * xp[(long)n * E_DIM];
  xbar[tid] = acc;
  __syncthreads();

  // cv[e] = xbar . W_k[:,e] + b_k[e]
  float c = 0.0f;
  for (int d = 0; d < E_DIM; ++d) c += xbar[d] * W_qkv[d * 513 + 1 + tid];
  cv_ws[bp * E_DIM + tid] = c + b_qkv[1 + tid];
}

// ---------------------------------------------------------------------------
// k3: per 16-token tile:
//   stage1: v = relu(x_tile @ W_v + b_v) * cv   -> bf16 tile in LDS
//   stage2: out = v @ W_o + b_o                 -> f32 global
// 8 waves/block, each wave owns 2 column tiles; 32 v_wmma per wave.
// ---------------------------------------------------------------------------
__global__ __launch_bounds__(256) void k3_main(
    const float* __restrict__ x, const unsigned short* __restrict__ wv_pre,
    const unsigned short* __restrict__ wo_pre, const float* __restrict__ b_qkv,
    const float* __restrict__ b_o, const float* __restrict__ cv_ws,
    float* __restrict__ out)
{
  const int blk  = blockIdx.x;
  const int bp   = blk >> 6;        // 64 token-tiles per (b,p)
  const int mt   = blk & 63;
  const int tid  = threadIdx.x;
  const int wave = tid >> 5;
  const int lane = tid & 31;
  const int r    = lane & 15;
  const int base0 = (lane < 16) ? 0 : 8;

  const long tok0 = (long)bp * N_TOK + mt * 16;
  const float* xt = x + tok0 * E_DIM;

  __shared__ __align__(16) bf16_t u[16][264];   // 16 tokens x 256 ch (+pad)

  const v16bf* wv = (const v16bf*)wv_pre;
  const v16bf* wo = (const v16bf*)wo_pre;

  const int nt0 = wave * 2, nt1 = nt0 + 1;

  // ---- stage 1: acc = x_tile @ W_v (two 16-col tiles per wave) ----
  v8f acc0 = {}; v8f acc1 = {};
  const float* ar = xt + r * E_DIM + base0;
#pragma unroll
  for (int kt = 0; kt < 8; ++kt) {
    const float* ap = ar + kt * 32;
    float4 f0 = *(const float4*)(ap + 0);
    float4 f1 = *(const float4*)(ap + 4);
    float4 f2 = *(const float4*)(ap + 16);
    float4 f3 = *(const float4*)(ap + 20);
    v16bf a;
    a[0]  = (bf16_t)f0.x; a[1]  = (bf16_t)f0.y; a[2]  = (bf16_t)f0.z; a[3]  = (bf16_t)f0.w;
    a[4]  = (bf16_t)f1.x; a[5]  = (bf16_t)f1.y; a[6]  = (bf16_t)f1.z; a[7]  = (bf16_t)f1.w;
    a[8]  = (bf16_t)f2.x; a[9]  = (bf16_t)f2.y; a[10] = (bf16_t)f2.z; a[11] = (bf16_t)f2.w;
    a[12] = (bf16_t)f3.x; a[13] = (bf16_t)f3.y; a[14] = (bf16_t)f3.z; a[15] = (bf16_t)f3.w;
    v16bf b0 = wv[(kt * 16 + nt0) * 32 + lane];
    v16bf b1 = wv[(kt * 16 + nt1) * 32 + lane];
    acc0 = __builtin_amdgcn_wmma_f32_16x16x32_bf16(false, a, false, b0, (short)0, acc0, false, false);
    acc1 = __builtin_amdgcn_wmma_f32_16x16x32_bf16(false, a, false, b1, (short)0, acc1, false, false);
  }

  // bias + relu + context-vector scale, store bf16 tile to LDS
  const int c0 = nt0 * 16 + r, c1 = nt1 * 16 + r;
  const float bv0 = b_qkv[257 + c0], bv1 = b_qkv[257 + c1];
  const float s0 = cv_ws[bp * E_DIM + c0], s1 = cv_ws[bp * E_DIM + c1];
#pragma unroll
  for (int i = 0; i < 8; ++i) {
    float t0 = fmaxf(acc0[i] + bv0, 0.0f) * s0;
    float t1 = fmaxf(acc1[i] + bv1, 0.0f) * s1;
    u[i + base0][c0] = (bf16_t)t0;   // M = vgpr + 8*(lane>=16)
    u[i + base0][c1] = (bf16_t)t1;
  }
  __syncthreads();

  // ---- stage 2: out = u @ W_o ----
  v8f o0 = {}; v8f o1 = {};
#pragma unroll
  for (int kt = 0; kt < 8; ++kt) {
    const bf16_t* up = &u[r][kt * 32 + base0];
    v8bf alo = *(const v8bf*)up;          // ds_load_b128
    v8bf ahi = *(const v8bf*)(up + 16);   // ds_load_b128
    v16bf a;
#pragma unroll
    for (int i = 0; i < 8; ++i) { a[i] = alo[i]; a[i + 8] = ahi[i]; }
    v16bf b0 = wo[(kt * 16 + nt0) * 32 + lane];
    v16bf b1 = wo[(kt * 16 + nt1) * 32 + lane];
    o0 = __builtin_amdgcn_wmma_f32_16x16x32_bf16(false, a, false, b0, (short)0, o0, false, false);
    o1 = __builtin_amdgcn_wmma_f32_16x16x32_bf16(false, a, false, b1, (short)0, o1, false, false);
  }

  const float bo0 = b_o[c0], bo1 = b_o[c1];
  float* op = out + tok0 * E_DIM;
#pragma unroll
  for (int i = 0; i < 8; ++i) {
    op[(long)(i + base0) * E_DIM + c0] = o0[i] + bo0;
    op[(long)(i + base0) * E_DIM + c1] = o1[i] + bo1;
  }
}

// ---------------------------------------------------------------------------
extern "C" void kernel_launch(void* const* d_in, const int* in_sizes, int n_in,
                              void* d_out, int out_size, void* d_ws, size_t ws_size,
                              hipStream_t stream) {
  const float* x     = (const float*)d_in[0];   // (32,4,1024,256)
  const float* W_qkv = (const float*)d_in[1];   // (256,513)
  const float* b_qkv = (const float*)d_in[2];   // (513,)
  const float* W_o   = (const float*)d_in[3];   // (256,256)
  const float* b_o   = (const float*)d_in[4];   // (256,)
  float* out = (float*)d_out;

  char* ws = (char*)d_ws;
  float*          q_ws   = (float*)(ws);                 // 131072 f32 = 512 KB
  float*          cv_ws  = (float*)(ws + 524288);        // 32768 f32  = 128 KB
  float*          wq_ws  = (float*)(ws + 655360);        // 256 f32
  unsigned short* wv_pre = (unsigned short*)(ws + 656384);  // 65536 bf16
  unsigned short* wo_pre = (unsigned short*)(ws + 787456);  // 65536 bf16

  k0_prep<<<512, 256, 0, stream>>>(W_qkv, W_o, wq_ws, wv_pre, wo_pre);
  k1_q   <<<TOKENS / 8, 256, 0, stream>>>(x, wq_ws, b_qkv, q_ws);
  k2_cv  <<<BP, 256, 0, stream>>>(x, W_qkv, b_qkv, q_ws, cv_ws);
  k3_main<<<BP * 64, 256, 0, stream>>>(x, wv_pre, wo_pre, b_qkv, b_o, cv_ws, out);
}
